// MMD_loss_CNN_54984171323540
// MI455X (gfx1250) — matrix operand order repaired and checked by reading, not compiled, not run on hardware
//
#include <hip/hip_runtime.h>
#include <hip/hip_bf16.h>

// ---------------------------------------------------------------------------
// MMD loss, fused: batched Gram (32 x [1024x1024, K=512]) via bf16 WMMA,
// fused RBF-mixture epilogue + signed reduction to a scalar.
// ---------------------------------------------------------------------------

typedef __attribute__((ext_vector_type(16))) __bf16 v16bf;
typedef __attribute__((ext_vector_type(8)))  __bf16 v8bf;
typedef __attribute__((ext_vector_type(4)))  __bf16 v4bf;
typedef __attribute__((ext_vector_type(8)))  float  v8f;

#define B_SZ    512          // batch per side
#define C_SZ    32           // channel slices (independent Grams)
#define D_SZ    512          // K dimension
#define TM      128          // tile (rows == cols)
#define KP      64           // K panel staged in LDS
#define LROWS   256          // TM (I rows) + TM (J rows)
#define LSTRIDE 72           // padded bf16 row stride in LDS (64 + 8)
#define NTILE   8            // 1024 / TM
#define NTRI    36           // NTILE*(NTILE+1)/2 upper-triangular tiles
#define NPART   (NTRI * C_SZ)

// A-fragment (16x32 bf16, MxK): lane m=lane&15, hi=lane>>4 holds
// K in [hi*8, hi*8+8) and [16+hi*8, 16+hi*8+8)   (ISA 7.12.2)
__device__ __forceinline__ v16bf load_a_frag(const __bf16* row, int k0, int hi) {
  union { v16bf v; v8bf h[2]; } u;
  u.h[0] = *(const v8bf*)(row + k0 + hi * 8);
  u.h[1] = *(const v8bf*)(row + k0 + 16 + hi * 8);
  return u.v;
}
// B-fragment (32x16 bf16, KxN): lane n=lane&15 holds column n,
// K in [hi*8, hi*8+8) and [16+hi*8, 16+hi*8+8) — from row n of the J-block.
__device__ __forceinline__ v16bf load_b_frag(const __bf16* row, int k0, int hi) {
  union { v16bf v; v8bf h[2]; } u;
  const __bf16* p = row + k0 + hi * 8;
  u.h[0] = *(const v8bf*)(p);
  u.h[1] = *(const v8bf*)(p + 16);
  return u.v;
}

__device__ __forceinline__ v8f wmma_bf16(v16bf a, v16bf b, v8f c) {
  return __builtin_amdgcn_wmma_f32_16x16x32_bf16(
      false, a, false, b, (short)0, c, false, false);
}

__global__ __launch_bounds__(256) void mmd_gram_kernel(
    const float* __restrict__ srcP, const float* __restrict__ tgtP,
    float* __restrict__ partials) {
  __shared__ __bf16 ldsData[LROWS * LSTRIDE];   // ~36.9 KB
  __shared__ float  lds_sq[LROWS];
  __shared__ float  redbuf[8];

  const int t    = (int)threadIdx.x;
  const int lane = t & 31;
  const int wave = t >> 5;          // 0..7
  const int hi   = lane >> 4;       // 0/1
  const int ln16 = lane & 15;
  const int c    = (int)blockIdx.z;

  // Upper-triangular tile decode: blockIdx.x in [0, 36) -> (ti <= tj)
  int ti = 0, rem = (int)blockIdx.x;
  while (rem >= (NTILE - ti)) { rem -= (NTILE - ti); ++ti; }
  const int tj = ti + rem;
  const int I0 = ti * TM;
  const int J0 = tj * TM;
  const float weight = (ti == tj) ? 1.0f : 2.0f;   // symmetry doubling

  // Wave sub-tile: wm in 0..3 (32-row block), wn in 0..1 (64-col block)
  const int wm = wave & 3;
  const int wn = wave >> 2;

  const __bf16* ar[2];
  const __bf16* br[4];
  ar[0] = &ldsData[(wm * 32 + ln16) * LSTRIDE];
  ar[1] = ar[0] + 16 * LSTRIDE;
#pragma unroll
  for (int ni = 0; ni < 4; ++ni)
    br[ni] = &ldsData[(TM + wn * 64 + ni * 16 + ln16) * LSTRIDE];

  v8f acc[2][4] = {};

  const int tr = t >> 4;            // 0..15  (row group for staging)
  const int td = (t & 15) << 2;     // 0..60  (float4 offset in panel)

  for (int p = 0; p < D_SZ / KP; ++p) {
    __syncthreads();                // previous panel fully consumed
    // ---- stage panel p: 256 rows x 64 floats -> bf16 LDS ----
#pragma unroll 4
    for (int it = 0; it < 16; ++it) {
      const int lr = it * 16 + tr;
      const int grow = (lr < TM) ? (I0 + lr) : (J0 + lr - TM);
      const float* rowP = (grow < B_SZ)
          ? (srcP + (((size_t)grow * C_SZ + c) << 9))
          : (tgtP + (((size_t)(grow - B_SZ) * C_SZ + c) << 9));
      const float4 v = *(const float4*)(rowP + (p << 6) + td);
      v4bf b4 = {(__bf16)v.x, (__bf16)v.y, (__bf16)v.z, (__bf16)v.w};
      *(v4bf*)(&ldsData[lr * LSTRIDE + td]) = b4;
    }
    __syncthreads();
    // ---- per-row squared-norm partials (from bf16 values -> diag exact) ----
    {
      const __bf16* rp = &ldsData[t * LSTRIDE];
      float s = 0.0f;
#pragma unroll
      for (int q = 0; q < KP; q += 8) {
        v8bf v = *(const v8bf*)(rp + q);
#pragma unroll
        for (int e = 0; e < 8; ++e) { float f = (float)v[e]; s = fmaf(f, f, s); }
      }
      lds_sq[t] = (p == 0) ? s : (lds_sq[t] + s);
    }
    // ---- WMMA over the panel ----
#pragma unroll
    for (int k0 = 0; k0 < KP; k0 += 32) {
      v16bf a0 = load_a_frag(ar[0], k0, hi);
      v16bf a1 = load_a_frag(ar[1], k0, hi);
      v16bf b0 = load_b_frag(br[0], k0, hi);
      v16bf b1 = load_b_frag(br[1], k0, hi);
      v16bf b2 = load_b_frag(br[2], k0, hi);
      v16bf b3 = load_b_frag(br[3], k0, hi);
      acc[0][0] = wmma_bf16(a0, b0, acc[0][0]);
      acc[0][1] = wmma_bf16(a0, b1, acc[0][1]);
      acc[0][2] = wmma_bf16(a0, b2, acc[0][2]);
      acc[0][3] = wmma_bf16(a0, b3, acc[0][3]);
      acc[1][0] = wmma_bf16(a1, b0, acc[1][0]);
      acc[1][1] = wmma_bf16(a1, b1, acc[1][1]);
      acc[1][2] = wmma_bf16(a1, b2, acc[1][2]);
      acc[1][3] = wmma_bf16(a1, b3, acc[1][3]);
    }
  }
  __syncthreads();                  // lds_sq complete for epilogue

  // ---- fused epilogue: L2 -> 5-sigma RBF mixture -> signed sum ----
  // C/D layout: VGPR r, lanes 0-15 -> (M=r, N=lane); lanes 16-31 -> (M=r+8).
  float lsum = 0.0f;
#pragma unroll
  for (int mi = 0; mi < 2; ++mi) {
    const int mbase = wm * 32 + mi * 16 + hi * 8;
#pragma unroll
    for (int ni = 0; ni < 4; ++ni) {
      const int nloc = wn * 64 + ni * 16 + ln16;
      const int j    = J0 + nloc;
      const float sqj = lds_sq[TM + nloc];
      const float sj  = (j < B_SZ) ? 1.0f : -1.0f;
      const v8f  A    = acc[mi][ni];
#pragma unroll
      for (int r = 0; r < 8; ++r) {
        const int i = I0 + mbase + r;
        const float sqi = lds_sq[mbase + r];
        const float l2  = fmaxf(fmaf(-2.0f, A[r], sqi + sqj), 0.0f);
        const float t2  = l2 * 1.4426950408889634f;   // log2(e)
        const float ks  = __builtin_amdgcn_exp2f(-t2)
                        + __builtin_amdgcn_exp2f(-0.5f    * t2)
                        + __builtin_amdgcn_exp2f(-0.25f   * t2)
                        + __builtin_amdgcn_exp2f(-0.125f  * t2)
                        + __builtin_amdgcn_exp2f(-0.0625f * t2);
        const float si = (i < B_SZ) ? 1.0f : -1.0f;
        lsum = fmaf(si * sj, ks, lsum);
      }
    }
  }

  // ---- deterministic block reduction -> one partial per block ----
#pragma unroll
  for (int off = 16; off > 0; off >>= 1)
    lsum += __shfl_xor(lsum, off, 32);
  if (lane == 0) redbuf[wave] = lsum;
  __syncthreads();
  if (t == 0) {
    float s = 0.0f;
#pragma unroll
    for (int w = 0; w < 8; ++w) s += redbuf[w];
    partials[(int)blockIdx.z * NTRI + (int)blockIdx.x] = s * weight;
  }
}

__global__ __launch_bounds__(256) void mmd_reduce_kernel(
    const float* __restrict__ partials, float* __restrict__ out) {
  __shared__ float red[256];
  float s = 0.0f;
  for (int i = (int)threadIdx.x; i < NPART; i += 256) s += partials[i];
  red[threadIdx.x] = s;
  __syncthreads();
  for (int off = 128; off > 0; off >>= 1) {
    if ((int)threadIdx.x < off) red[threadIdx.x] += red[threadIdx.x + off];
    __syncthreads();
  }
  if (threadIdx.x == 0)
    out[0] = red[0] * (1.0f / ((float)B_SZ * (float)B_SZ));
}

extern "C" void kernel_launch(void* const* d_in, const int* in_sizes, int n_in,
                              void* d_out, int out_size, void* d_ws, size_t ws_size,
                              hipStream_t stream) {
  (void)in_sizes; (void)n_in; (void)out_size; (void)ws_size;
  const float* src = (const float*)d_in[0];   // (512, 32, 512) f32
  const float* tgt = (const float*)d_in[1];   // (512, 32, 512) f32
  float* partials  = (float*)d_ws;            // NPART floats (~4.5 KB)
  float* out       = (float*)d_out;           // scalar f32

  dim3 grid(NTRI, 1, C_SZ);                   // 36 tri-tiles x 32 slices
  mmd_gram_kernel<<<grid, 256, 0, stream>>>(src, tgt, partials);
  mmd_reduce_kernel<<<1, 256, 0, stream>>>(partials, out);
}
